// MPNNReg_33406255628688
// MI455X (gfx1250) — matrix operand order repaired and checked by reading, not compiled, not run on hardware
//
#include <hip/hip_runtime.h>
#include <hip/hip_bf16.h>

// ---------------------------------------------------------------------------
// MPNN regression on gfx1250.
//   h  = relu(x @ W_in + b_in)
//   2x: p = relu(h @ msg_W + msg_b)          (gather commutes with matmul!)
//       aggr[dst] += p[src]                  (f32 atomics, L2-resident)
//       h = relu(concat(h,aggr) @ upd_W + b) (K=1024 GEMM over concat buffer)
//   out = h @ W_out + b_out
// GEMMs: v_wmma_f32_16x16x32_bf16, wave tile 64x64 (16 accumulators),
// WG = 4 waves = 128x128 tile.
// ---------------------------------------------------------------------------

typedef __attribute__((ext_vector_type(16))) __bf16 v16bf;
typedef __attribute__((ext_vector_type(8)))  __bf16 v8bf;
typedef __attribute__((ext_vector_type(8)))  float  v8f;

#define N_NODES 10000
#define N_EDGES 160000
#define IN_CH   256
#define HIDDEN  512
#define M_PAD   10112   // 79 * 128, multiple of the 128-row workgroup tile

// ---------------- fragment loaders (documented CDNA5 VGPR layouts) ----------

// A: 16x32 bf16, row-major source, stride lda.
// lane<16 holds M=lane, K-chunks [0..8) and [16..24); lane>=16 holds
// M=lane-16, K-chunks [8..16) and [24..32).
__device__ __forceinline__ v16bf load_a_frag(const __bf16* __restrict__ A,
                                             int lda, int m16, int k0,
                                             int r, int half) {
  const __bf16* base = A + (size_t)(m16 + r) * lda + k0 + half * 8;
  v8bf lo = *(const v8bf*)base;         // K = k0 + half*8 .. +8
  v8bf hi = *(const v8bf*)(base + 16);  // K = k0 + 16 + half*8 .. +8
  v16bf f;
#pragma unroll
  for (int i = 0; i < 8; ++i) { f[i] = lo[i]; f[i + 8] = hi[i]; }
  return f;
}

// B: 32x16 bf16 from pre-transposed weights Wt[N][K] (row-major, stride ldb).
// lane<16 holds N=lane, K-chunks [0..8) and [16..24); lane>=16 holds
// N=lane-16, K-chunks [8..16) and [24..32).
__device__ __forceinline__ v16bf load_b_frag(const __bf16* __restrict__ Bt,
                                             int ldb, int n16, int k0,
                                             int r, int half) {
  const __bf16* base = Bt + (size_t)(n16 + r) * ldb + k0 + half * 8;
  v8bf lo = *(const v8bf*)base;
  v8bf hi = *(const v8bf*)(base + 16);
  v16bf f;
#pragma unroll
  for (int i = 0; i < 8; ++i) { f[i] = lo[i]; f[i + 8] = hi[i]; }
  return f;
}

// ---------------- GEMM: C = relu(A(bf16) @ Wt^T(bf16) + bias) --------------
// Templated on output type -> single unconditional store path per instance.
// grid = (N/128, M_PAD/128), block = 128 threads (4 waves, 2x2).
template <typename OutT>
__global__ __launch_bounds__(128)
void k_gemm_bias_relu(const __bf16* __restrict__ A, int lda,
                      const __bf16* __restrict__ Bt, int ldb,
                      const float* __restrict__ bias,
                      OutT* __restrict__ C, int ldc, int K) {
  const int lane = threadIdx.x & 31;
  const int wave = threadIdx.x >> 5;
  const int r    = lane & 15;
  const int half = lane >> 4;
  const int m0 = blockIdx.y * 128 + (wave >> 1) * 64;
  const int n0 = blockIdx.x * 128 + (wave & 1) * 64;

  v8f acc[4][4];
#pragma unroll
  for (int mi = 0; mi < 4; ++mi)
#pragma unroll
    for (int ni = 0; ni < 4; ++ni)
#pragma unroll
      for (int g = 0; g < 8; ++g) acc[mi][ni][g] = 0.0f;

  for (int k0 = 0; k0 < K; k0 += 32) {
    v16bf af[4], bf[4];
#pragma unroll
    for (int mi = 0; mi < 4; ++mi)
      af[mi] = load_a_frag(A, lda, m0 + mi * 16, k0, r, half);
#pragma unroll
    for (int ni = 0; ni < 4; ++ni)
      bf[ni] = load_b_frag(Bt, ldb, n0 + ni * 16, k0, r, half);
#pragma unroll
    for (int mi = 0; mi < 4; ++mi)
#pragma unroll
      for (int ni = 0; ni < 4; ++ni)
        acc[mi][ni] = __builtin_amdgcn_wmma_f32_16x16x32_bf16(
            false, af[mi], false, bf[ni], (short)0, acc[mi][ni], false, false);
  }

  // epilogue: bias + relu + store. element g = (row m0+mi*16+half*8+g, col n0+ni*16+r)
#pragma unroll
  for (int ni = 0; ni < 4; ++ni) {
    const float bv = bias[n0 + ni * 16 + r];
#pragma unroll
    for (int mi = 0; mi < 4; ++mi) {
      OutT* cp = C + (size_t)(m0 + mi * 16 + half * 8) * ldc + n0 + ni * 16 + r;
#pragma unroll
      for (int g = 0; g < 8; ++g) {
        cp[(size_t)g * ldc] = (OutT)fmaxf(acc[mi][ni][g] + bv, 0.0f);
      }
    }
  }
}

// ---------------- helper kernels -------------------------------------------

// f32 -> bf16 with zero row padding (same column count src/dst).
__global__ void k_cast_pad(const float* __restrict__ src, __bf16* __restrict__ dst,
                           int rows, int cols, int rows_pad) {
  size_t total = (size_t)rows_pad * cols;
  size_t stride = (size_t)gridDim.x * blockDim.x;
  for (size_t i = (size_t)blockIdx.x * blockDim.x + threadIdx.x; i < total; i += stride) {
    size_t row = i / cols;
    dst[i] = (row < (size_t)rows) ? (__bf16)src[i] : (__bf16)0.0f;
  }
}

// W[K][N] f32 (row-major) -> Wt[N][K] bf16 (row-major).
__global__ void k_transpose_cast(const float* __restrict__ W, __bf16* __restrict__ Wt,
                                 int K, int N) {
  size_t total = (size_t)K * N;
  size_t stride = (size_t)gridDim.x * blockDim.x;
  for (size_t i = (size_t)blockIdx.x * blockDim.x + threadIdx.x; i < total; i += stride) {
    size_t n = i / K, k = i % K;
    Wt[i] = (__bf16)W[k * (size_t)N + n];
  }
}

__global__ void k_zero_f32(float* __restrict__ p, size_t n) {
  size_t stride = (size_t)gridDim.x * blockDim.x;
  for (size_t i = (size_t)blockIdx.x * blockDim.x + threadIdx.x; i < n; i += stride)
    p[i] = 0.0f;
}

// One wave per edge: aggr[dst] += p[src]  (512 f32 channels, 16 per lane).
__global__ void k_edge_scatter(const float* __restrict__ p,
                               const int* __restrict__ ei, int E,
                               float* __restrict__ aggr) {
  const int lane  = threadIdx.x & 31;
  const int warp  = (blockIdx.x * blockDim.x + threadIdx.x) >> 5;
  const int nwarp = (gridDim.x * blockDim.x) >> 5;
  for (int e = warp; e < E; e += nwarp) {
    const int s = ei[e];          // src node
    const int d = ei[E + e];      // dst node
    const float4* ps = (const float4*)(p + (size_t)s * HIDDEN + lane * 16);
    float* pd = aggr + (size_t)d * HIDDEN + lane * 16;
#pragma unroll
    for (int i = 0; i < 4; ++i) {
      float4 v = ps[i];
      atomicAdd(pd + i * 4 + 0, v.x);
      atomicAdd(pd + i * 4 + 1, v.y);
      atomicAdd(pd + i * 4 + 2, v.z);
      atomicAdd(pd + i * 4 + 3, v.w);
    }
  }
}

// aggr[M_PAD][512] f32 -> cols [512,1024) of h (bf16, stride 1024).
__global__ void k_cast_concat(const float* __restrict__ aggr, __bf16* __restrict__ h,
                              size_t rows) {
  size_t total = rows * HIDDEN;
  size_t stride = (size_t)gridDim.x * blockDim.x;
  for (size_t i = (size_t)blockIdx.x * blockDim.x + threadIdx.x; i < total; i += stride) {
    size_t row = i >> 9;          // /512
    int c = (int)(i & 511);
    h[row * 1024 + HIDDEN + c] = (__bf16)aggr[i];
  }
}

// out[i] = h[i,:512] . W_out + b_out   (h stride 1024, bf16)
__global__ void k_out_proj(const __bf16* __restrict__ h,
                           const float* __restrict__ Wo,
                           const float* __restrict__ bo,
                           float* __restrict__ out, int n) {
  int i = blockIdx.x * blockDim.x + threadIdx.x;
  if (i >= n) return;
  const __bf16* hr = h + (size_t)i * 1024;
  float s = 0.0f;
#pragma unroll 8
  for (int k = 0; k < HIDDEN; ++k) s += (float)hr[k] * Wo[k];
  out[i] = s + bo[0];
}

// ---------------- launcher --------------------------------------------------

extern "C" void kernel_launch(void* const* d_in, const int* in_sizes, int n_in,
                              void* d_out, int out_size, void* d_ws, size_t ws_size,
                              hipStream_t stream) {
  const float* x     = (const float*)d_in[0];
  const int*   ei    = (const int*)d_in[1];   // [2, E]: row0=src, row1=dst
  const float* W_in  = (const float*)d_in[2]; // [256, 512]
  const float* b_in  = (const float*)d_in[3]; // [512]
  const float* msg_W = (const float*)d_in[4]; // [2, 512, 512]
  const float* msg_b = (const float*)d_in[5]; // [2, 512]
  const float* upd_W = (const float*)d_in[6]; // [2, 1024, 512]
  const float* upd_b = (const float*)d_in[7]; // [2, 512]
  const float* W_out = (const float*)d_in[8]; // [512, 1]
  const float* b_out = (const float*)d_in[9]; // [1]
  float* out = (float*)d_out;
  const int E = in_sizes[1] / 2;

  // ---- carve workspace (256B aligned chunks) ----
  char* w = (char*)d_ws;
  auto carve = [&](size_t bytes) -> void* {
    void* ptr = (void*)w;
    w += (bytes + 255) & ~(size_t)255;
    return ptr;
  };
  __bf16* xb    = (__bf16*)carve((size_t)M_PAD * IN_CH * 2);
  __bf16* WtIn  = (__bf16*)carve((size_t)HIDDEN * IN_CH * 2);
  __bf16* msgWt = (__bf16*)carve((size_t)2 * HIDDEN * HIDDEN * 2);
  __bf16* updWt = (__bf16*)carve((size_t)2 * HIDDEN * 1024 * 2);
  __bf16* hA    = (__bf16*)carve((size_t)M_PAD * 1024 * 2);
  __bf16* hB    = (__bf16*)carve((size_t)M_PAD * 1024 * 2);
  float*  p     = (float*)carve((size_t)M_PAD * HIDDEN * 4);
  float*  aggr  = (float*)carve((size_t)M_PAD * HIDDEN * 4);
  (void)ws_size; (void)n_in; (void)out_size;

  // ---- weight prep: transpose + cast to bf16 ----
  k_transpose_cast<<<512, 256, 0, stream>>>(W_in, WtIn, IN_CH, HIDDEN);
  for (int l = 0; l < 2; ++l) {
    k_transpose_cast<<<1024, 256, 0, stream>>>(
        msg_W + (size_t)l * HIDDEN * HIDDEN, msgWt + (size_t)l * HIDDEN * HIDDEN,
        HIDDEN, HIDDEN);
    k_transpose_cast<<<2048, 256, 0, stream>>>(
        upd_W + (size_t)l * 1024 * HIDDEN, updWt + (size_t)l * HIDDEN * 1024,
        1024, HIDDEN);
  }
  k_cast_pad<<<2048, 256, 0, stream>>>(x, xb, N_NODES, IN_CH, M_PAD);

  const dim3 ggrid(HIDDEN / 128, M_PAD / 128);

  // h = relu(x @ W_in + b_in)  -> hA cols [0,512)
  k_gemm_bias_relu<__bf16><<<ggrid, 128, 0, stream>>>(xb, IN_CH, WtIn, IN_CH,
                                                      b_in, hA, 1024, IN_CH);

  __bf16* cur = hA;
  __bf16* nxt = hB;
  for (int l = 0; l < 2; ++l) {
    // p = relu(h @ msg_W[l] + msg_b[l])   (relu commutes with the edge gather)
    k_gemm_bias_relu<float><<<ggrid, 128, 0, stream>>>(
        cur, 1024, msgWt + (size_t)l * HIDDEN * HIDDEN, HIDDEN,
        msg_b + (size_t)l * HIDDEN, p, HIDDEN, HIDDEN);
    // aggr = scatter_add(p[src] -> dst)
    k_zero_f32<<<2048, 256, 0, stream>>>(aggr, (size_t)M_PAD * HIDDEN);
    k_edge_scatter<<<2048, 256, 0, stream>>>(p, ei, E, aggr);
    // concat: cur cols [512,1024) = bf16(aggr)
    k_cast_concat<<<2048, 256, 0, stream>>>(aggr, cur, (size_t)M_PAD);
    // h = relu(concat(h,aggr) @ upd_W[l] + upd_b[l]) -> nxt cols [0,512)
    k_gemm_bias_relu<__bf16><<<ggrid, 128, 0, stream>>>(
        cur, 1024, updWt + (size_t)l * HIDDEN * 1024, 1024,
        upd_b + (size_t)l * HIDDEN, nxt, 1024, 1024);
    __bf16* t = cur; cur = nxt; nxt = t;
  }

  // out = h @ W_out + b_out
  k_out_proj<<<(N_NODES + 255) / 256, 256, 0, stream>>>(cur, W_out, b_out, out, N_NODES);
}
